// LSTM_56238301774390
// MI455X (gfx1250) — matrix operand order repaired and checked by reading, not compiled, not run on hardware
//
#include <hip/hip_runtime.h>

// ---------------- types ----------------
typedef __bf16 v16bf __attribute__((ext_vector_type(16)));
typedef float  v8f   __attribute__((ext_vector_type(8)));
typedef unsigned int u32x4 __attribute__((ext_vector_type(4)));
typedef unsigned int u32x8 __attribute__((ext_vector_type(8)));

union Frag {
    v16bf v;
    uint4 q[2];
};

// ---------------- problem constants ----------------
#define SEQ  512
#define BAT  64
#define IDIM 1024
#define HDIM 1024
#define GDIM 4096          // 4*H
#define KTOT 2048          // I + H
#define KC   128           // K chunk staged in LDS
#define NCHUNK (KTOT / KC) // 16 (chunks 0..7 from x, 8..15 from h)
#define AST  136           // As row stride (bf16): 128 + 8 pad (matches TDM pad descriptor)
#define CST  68            // Act row stride (f32)
#define XCH  16            // timesteps per x-conversion batch

// ---------------- helpers ----------------
__device__ __forceinline__ unsigned short f2bf(float f) {
    unsigned int u = __float_as_uint(f);
    unsigned int r = u + 0x7FFFu + ((u >> 16) & 1u);   // round-to-nearest-even
    return (unsigned short)(r >> 16);
}
__device__ __forceinline__ float fast_sigmoid(float x) {
    return 1.0f / (1.0f + __expf(-x));
}
__device__ __forceinline__ float fast_tanh(float x) {
    float e = __expf(-2.0f * fabsf(x));
    float r = (1.0f - e) / (1.0f + e);
    return copysignf(r, x);
}

// ---- Tensor Data Mover: DMA a 64-row x 128-col bf16 tile (row stride 1024 elems)
// from global memory into LDS, inserting 16B of padding after every 256B row so the
// LDS image has row stride AST=136 bf16 (bank-conflict-free for the WMMA A loads).
// D# per CDNA5 ISA 8.3/8.4: group0 = {count/lds_addr/global_addr/type}, group1 =
// {data_size=2B, pad_enable, pad_interval=64dw, pad_amount=4dw, dims/strides/tile}.
__device__ __forceinline__ void tdm_load_tile(const unsigned short* src, unsigned lds_byte_off) {
    unsigned long long ga = (unsigned long long)(uintptr_t)src;
    u32x4 g0;
    g0[0] = 1u;                                              // count=1, user descriptor
    g0[1] = lds_byte_off;                                    // LDS dest byte address
    g0[2] = (unsigned)ga;                                    // global_addr[31:0]
    g0[3] = (unsigned)((ga >> 32) & 0x01FFFFFFull) | (2u << 30); // addr[56:32] | type=2
    u32x8 g1;
    g1[0] = (1u << 16)        // data_size = 1 -> 2 bytes
          | (1u << 20)        // pad_enable
          | (5u << 22)        // pad_interval code 5 -> 64 DWORDs (= one 256B row)
          | (3u << 25);       // pad_amount  code 3 -> 4 DWORDs (= 16B = 8 bf16)
    g1[1] = (1024u & 0xFFFFu) << 16;          // tensor_dim0[15:0]  (row length 1024)
    g1[2] = (1024u >> 16) | (64u << 16);      // tensor_dim0[31:16] | tensor_dim1[15:0]=64
    g1[3] = (128u << 16);                     // tensor_dim1[31:16] | tile_dim0=128
    g1[4] = 64u;                              // tile_dim1=64 | tile_dim2=0
    g1[5] = 1024u;                            // tensor_dim0_stride[31:0] = 1024 elems
    g1[6] = 0u;                               // stride0 hi | tensor_dim1_stride lo (2D: 0)
    g1[7] = 0u;
    asm volatile("tensor_load_to_lds %0, %1" :: "s"(g0), "s"(g1) : "memory");
}

// ---------------- phase 0: pack [Wi|Wh] -> bf16 with gate-interleaved rows ----------------
__global__ __launch_bounds__(256) void lstm_pack(const float* __restrict__ Wi,
                                                 const float* __restrict__ Wh,
                                                 const float* __restrict__ bi,
                                                 const float* __restrict__ bh,
                                                 unsigned short* __restrict__ Wp,
                                                 float* __restrict__ bsum) {
    int idx = blockIdx.x * 256 + threadIdx.x;     // 4096*2048/8 threads
    int p   = idx >> 8;
    int k0  = (idx & 255) * 8;
    int wg  = p >> 6;
    int r   = p & 63;
    int q   = r >> 4;
    int jj  = r & 15;
    int g   = q * 1024 + wg * 16 + jj;
    const float* src = (k0 < IDIM) ? (Wi + (size_t)g * IDIM + k0)
                                   : (Wh + (size_t)g * HDIM + (k0 - IDIM));
    unsigned short* dst = Wp + (size_t)p * KTOT + k0;
#pragma unroll
    for (int u = 0; u < 8; ++u) dst[u] = f2bf(src[u]);
    if (k0 == 0) bsum[p] = bi[g] + bh[g];
}

// ---------------- phase 0: init h (bf16) and c (f32) state ----------------
__global__ __launch_bounds__(256) void lstm_init(const float* __restrict__ h0,
                                                 const float* __restrict__ c0,
                                                 unsigned short* __restrict__ hbuf,
                                                 float* __restrict__ cbuf) {
    int i = blockIdx.x * 256 + threadIdx.x;
    hbuf[i] = f2bf(h0[i]);
    cbuf[i] = c0[i];
}

// ---------------- x fp32 -> bf16 staging for XCH timesteps ----------------
__global__ __launch_bounds__(256) void lstm_xconv(const float* __restrict__ xc,
                                                  unsigned short* __restrict__ xs) {
    int i = (blockIdx.x * 256 + threadIdx.x) * 8;   // XCH*BAT*IDIM elems total
    float4 a = *(const float4*)(xc + i);
    float4 b = *(const float4*)(xc + i + 4);
    uint4 o;
    o.x = (unsigned)f2bf(a.x) | ((unsigned)f2bf(a.y) << 16);
    o.y = (unsigned)f2bf(a.z) | ((unsigned)f2bf(a.w) << 16);
    o.z = (unsigned)f2bf(b.x) | ((unsigned)f2bf(b.y) << 16);
    o.w = (unsigned)f2bf(b.z) | ((unsigned)f2bf(b.w) << 16);
    *(uint4*)(xs + i) = o;
}

// ---------------- recurrent step: pre = [x_t|h] @ Wp^T + bsum ; gates ; c,h update -----
// grid = 64 WGs (each owns 16 hidden indices: 64 packed cols = 4 gates x 16)
// block = 256 threads = 8 waves. Wave w: M-half (w>>2)*32, col group (w&3)*16.
// A panel is DMA'd by the Tensor Data Mover into double-buffered LDS, overlapping WMMA.
__global__ __launch_bounds__(256) void lstm_step(const unsigned short* __restrict__ xs,
                                                 const unsigned short* __restrict__ Wp,
                                                 const float* __restrict__ bsum,
                                                 const unsigned short* __restrict__ hin,
                                                 float* __restrict__ cbuf,
                                                 unsigned short* __restrict__ hout,
                                                 float* __restrict__ out_t) {
    __shared__ __align__(16) unsigned short As[2][BAT * AST];  // 2 x (64 x 128 bf16 +pad)
    __shared__ float Act[BAT * CST];

    const int tid  = threadIdx.x;
    const int wg   = blockIdx.x;        // 0..63
    const int w    = tid >> 5;          // wave 0..7 (wave32)
    const int l    = tid & 31;
    const int ln   = l & 15;
    const int half = l >> 4;
    const int nq   = w & 3;             // gate quadrant: 0=i 1=f 2=o 3=g
    const int m0   = (w >> 2) * 32;
    const int ncol = nq * 16 + ln;

    v8f acc0 = {};
    v8f acc1 = {};

    const unsigned short* wrow = Wp + (size_t)(wg * 64 + ncol) * KTOT;

    // chunk c source: c<8 -> staged x_t (bf16), else -> previous h (bf16)
    auto srcp = [&](int c) -> const unsigned short* {
        return (c < 8) ? (xs + c * KC) : (hin + (c - 8) * KC);
    };

    // prologue: DMA chunk 0 into buffer 0 (wave 0 issues; TDM ignores EXEC)
    if (w == 0) tdm_load_tile(srcp(0), (unsigned)(uintptr_t)&As[0][0]);

    for (int c = 0; c < NCHUNK; ++c) {
        if (w == 0) {
            if (c + 1 < NCHUNK) {
                tdm_load_tile(srcp(c + 1), (unsigned)(uintptr_t)&As[(c + 1) & 1][0]);
                __builtin_amdgcn_s_wait_tensorcnt(1);   // chunk c complete (in-order)
            } else {
                __builtin_amdgcn_s_wait_tensorcnt(0);
            }
        }
        __syncthreads();                                // panel c visible to all waves

        const unsigned short* Ab = &As[c & 1][0];
        const int kb = c * KC;
#pragma unroll
        for (int ks = 0; ks < KC; ks += 32) {
            Frag fb;  // B 32x16: lane ln = N = packed weight row, contiguous 16 K per half
            {
                const unsigned short* bp = wrow + kb + ks + half * 16;
                fb.q[0] = *(const uint4*)(bp);
                fb.q[1] = *(const uint4*)(bp + 8);
            }
            Frag fa0, fa1;  // A 16x32: lane<16 K{0..15}, lane>=16 K{16..31}
            {
                const unsigned short* ap0 = &Ab[(m0 + ln) * AST + ks + half * 16];
                fa0.q[0] = *(const uint4*)(ap0);
                fa0.q[1] = *(const uint4*)(ap0 + 8);
                const unsigned short* ap1 = &Ab[(m0 + 16 + ln) * AST + ks + half * 16];
                fa1.q[0] = *(const uint4*)(ap1);
                fa1.q[1] = *(const uint4*)(ap1 + 8);
            }
            acc0 = __builtin_amdgcn_wmma_f32_16x16x32_bf16(false, fa0.v, false, fb.v,
                                                           (short)0, acc0, false, false);
            acc1 = __builtin_amdgcn_wmma_f32_16x16x32_bf16(false, fa1.v, false, fb.v,
                                                           (short)0, acc1, false, false);
        }
        __syncthreads();   // all waves done reading buffer (c&1) before it is re-DMA'd
    }

    // ---- bias + per-gate activation, stage to LDS ----
    const float bs = bsum[wg * 64 + ncol];
#pragma unroll
    for (int r = 0; r < 8; ++r) {
        float v0 = acc0[r] + bs;
        float v1 = acc1[r] + bs;
        if (nq < 3) { v0 = fast_sigmoid(v0); v1 = fast_sigmoid(v1); }
        else        { v0 = fast_tanh(v0);    v1 = fast_tanh(v1);    }
        int mr0 = m0 + half * 8 + r;
        int mr1 = m0 + 16 + half * 8 + r;
        Act[mr0 * CST + ncol] = v0;
        Act[mr1 * CST + ncol] = v1;
    }
    __syncthreads();

    // ---- LSTM cell update: 64 batches x 16 hidden indices per WG ----
#pragma unroll
    for (int u = 0; u < 4; ++u) {
        int e  = u * 256 + tid;
        int b  = e >> 4;
        int jj = e & 15;
        float iv = Act[b * CST + jj];
        float fv = Act[b * CST + 16 + jj];
        float ov = Act[b * CST + 32 + jj];
        float gv = Act[b * CST + 48 + jj];
        int j = wg * 16 + jj;
        float c_old = cbuf[b * HDIM + j];
        float c_new = c_old * fv + iv * gv;
        float h     = ov * fast_tanh(c_new);
        cbuf[b * HDIM + j]  = c_new;
        hout[b * HDIM + j]  = f2bf(h);
        out_t[b * HDIM + j] = h;
    }
}

// ---------------- finalize ----------------
__global__ __launch_bounds__(256) void lstm_final(const float* __restrict__ last_h,
                                                  const float* __restrict__ cbuf,
                                                  float* __restrict__ hf,
                                                  float* __restrict__ cf) {
    int i = blockIdx.x * 256 + threadIdx.x;
    hf[i] = last_h[i];
    cf[i] = cbuf[i];
}

extern "C" void kernel_launch(void* const* d_in, const int* in_sizes, int n_in,
                              void* d_out, int out_size, void* d_ws, size_t ws_size,
                              hipStream_t stream) {
    (void)in_sizes; (void)n_in; (void)out_size; (void)ws_size;
    const float* x  = (const float*)d_in[0];
    const float* h0 = (const float*)d_in[1];
    const float* c0 = (const float*)d_in[2];
    const float* Wi = (const float*)d_in[3];
    const float* bi = (const float*)d_in[4];
    const float* Wh = (const float*)d_in[5];
    const float* bh = (const float*)d_in[6];
    float* out = (float*)d_out;

    // workspace carve (~18.6 MB)
    char* ws = (char*)d_ws;
    unsigned short* Wp     = (unsigned short*)ws; ws += (size_t)GDIM * KTOT * 2;   // 16 MB
    float*          bsum   = (float*)ws;          ws += (size_t)GDIM * 4;
    float*          cbuf   = (float*)ws;          ws += (size_t)BAT * HDIM * 4;
    unsigned short* hb0    = (unsigned short*)ws; ws += (size_t)BAT * HDIM * 2;
    unsigned short* hb1    = (unsigned short*)ws; ws += (size_t)BAT * HDIM * 2;
    unsigned short* xstage = (unsigned short*)ws; ws += (size_t)XCH * BAT * IDIM * 2; // 2 MB

    // phase 0
    lstm_pack<<<(GDIM * KTOT / 8) / 256, 256, 0, stream>>>(Wi, Wh, bi, bh, Wp, bsum);
    lstm_init<<<(BAT * HDIM) / 256, 256, 0, stream>>>(h0, c0, hb0, cbuf);

    // scan: every XCH steps convert an x slice to bf16 (parallel), then XCH fused steps.
    for (int t0 = 0; t0 < SEQ; t0 += XCH) {
        lstm_xconv<<<(XCH * BAT * IDIM / 8) / 256, 256, 0, stream>>>(
            x + (size_t)t0 * BAT * IDIM, xstage);
        for (int t = t0; t < t0 + XCH; ++t) {
            const unsigned short* hin = (t & 1) ? hb1 : hb0;
            unsigned short*      hout = (t & 1) ? hb0 : hb1;
            lstm_step<<<64, 256, 0, stream>>>(
                xstage + (size_t)(t - t0) * BAT * IDIM, Wp, bsum,
                hin, cbuf, hout, out + (size_t)t * BAT * HDIM);
        }
    }

    // tail: h_f, c_f
    float* hf = out + (size_t)SEQ * BAT * HDIM;
    float* cf = hf + (size_t)BAT * HDIM;
    lstm_final<<<(BAT * HDIM) / 256, 256, 0, stream>>>(out + (size_t)(SEQ - 1) * BAT * HDIM,
                                                       cbuf, hf, cf);
}
